// RNNDecoder_16569983828507
// MI455X (gfx1250) — matrix operand order repaired and checked by reading, not compiled, not run on hardware
//
#include <hip/hip_runtime.h>
#include <cstdint>
#include <cstddef>

// ---------------------------------------------------------------------------
// Problem constants (from reference)
// ---------------------------------------------------------------------------
#define BB   64
#define TT   600
#define UU   80
#define EE   256
#define HH   400
#define MAA  10
#define OUTN 121

#define LD_A0 288    // layer0 input  (3+256=259  -> pad 288 = 9*32)
#define LD_H  416    // hidden        (400        -> pad 416 = 13*32)
#define LD_X  672    // layer1/2 in   (3+400+256  -> pad 672 = 21*32)
#define LD_F  1216   // feats         (1200       -> pad 1216 = 38*32)
#define G4    1600   // 4*H
#define MROWS (TT*BB)   // 38400

typedef __attribute__((ext_vector_type(16))) __bf16 v16bf;
typedef __attribute__((ext_vector_type(8)))  float  v8f;

// ---------------------------------------------------------------------------
// WMMA helpers (CDNA5 bf16 16x16x32, f32 accumulate)
// ---------------------------------------------------------------------------
__device__ inline v8f wmma_bf16(v16bf a, v16bf b, v8f c) {
    return __builtin_amdgcn_wmma_f32_16x16x32_bf16(
        /*neg_a=*/false, a, /*neg_b=*/false, b,
        /*c_mod=*/(short)0, c, /*reuse_a=*/false, /*reuse_b=*/false);
}

// A fragment: 16x32 bf16, source row-major with leading dim `ld` (elements).
// ISA layout: lane l: M = l&15, half = l>>4; VGPR v: K = (v<4?0:16)+half*8+2*(v&3)
__device__ inline v16bf load_a_frag(const __bf16* src, int ld, int kBase, int lane) {
    union { v16bf v; uint32_t u[8]; } f;
    const int row = lane & 15;
    const int hlf = lane >> 4;
    const __bf16* p = src + (size_t)row * ld + kBase + hlf * 8;
#pragma unroll
    for (int v = 0; v < 8; ++v) {
        const int kOff = (v < 4 ? 0 : 16) + ((v & 3) * 2);
        f.u[v] = *(const uint32_t*)(p + kOff);
    }
    return f.v;
}

// B fragment: 32x16 bf16 from row-major Wt[Kpad][ldn] (i.e. W pre-transposed).
// ISA layout: lane l: N = l&15, half = l>>4; element s: K = (s<8?0:16)+half*8+(s&7)
__device__ inline v16bf load_b_frag(const __bf16* Wt, int ldn, int kBase, int nBase, int lane) {
    v16bf f;
    const __bf16* p = Wt + (size_t)(kBase + (lane >> 4) * 8) * ldn + nBase + (lane & 15);
#pragma unroll
    for (int s = 0; s < 16; ++s) f[s] = p[(size_t)((s < 8 ? 0 : 16) + (s & 7)) * ldn];
    return f;
}

__device__ inline float sigmf(float x) { return 1.0f / (1.0f + __expf(-x)); }

// ---------------------------------------------------------------------------
// Prep kernels: weight transpose f32[N][K] -> bf16[Kpad][Npad], bias fuse, pads
// ---------------------------------------------------------------------------
__global__ void transpose_bf16_kernel(const float* __restrict__ W, __bf16* __restrict__ Wt,
                                      int rows, int cols, int kpad, int npad) {
    int idx = blockIdx.x * blockDim.x + threadIdx.x;
    if (idx >= kpad * npad) return;
    int k = idx / npad, n = idx % npad;
    float v = (k < cols && n < rows) ? W[(size_t)n * cols + k] : 0.0f;
    Wt[(size_t)k * npad + n] = (__bf16)v;
}

__global__ void bias_fuse_kernel(const float* a, const float* b, float* o, int len, int lenpad) {
    int i = blockIdx.x * blockDim.x + threadIdx.x;
    if (i >= lenpad) return;
    o[i] = (i < len) ? (a[i] + (b ? b[i] : 0.0f)) : 0.0f;
}

__global__ void zeropad_kernel(__bf16* buf, int ld, int c0, int c1, int rows) {
    int total = rows * (c1 - c0);
    for (int i = blockIdx.x * blockDim.x + threadIdx.x; i < total; i += gridDim.x * blockDim.x) {
        int r = i / (c1 - c0), c = c0 + i % (c1 - c0);
        buf[(size_t)r * ld + c] = (__bf16)0.0f;
    }
}

// ---------------------------------------------------------------------------
// Layer-0 sequential scan + Graves attention. One persistent workgroup of
// 16 waves. nu-major mapping: a wave owns one 16-unit column group (all 4
// gates, all 4 batch tiles), so each weight element crosses L2 exactly once
// per step (4x less traffic than batch-major). c lives in VGPRs; h is
// double-buffered bf16 in LDS. Emits X1 = [x|h0|w|pad] and F[:,0:400]=h0.
// ---------------------------------------------------------------------------
__global__ void __launch_bounds__(512) scan0_kernel(
    const float* __restrict__ strokes, const float* __restrict__ context,
    const float* __restrict__ cmask,
    const __bf16* __restrict__ Wih0t, const __bf16* __restrict__ Whh0t,
    const float* __restrict__ bsum0,
    const float* __restrict__ W_att, const float* __restrict__ b_att,
    __bf16* __restrict__ X1, __bf16* __restrict__ F) {

    __shared__ __bf16 aT[BB][LD_A0];          // [x | w] bf16, padded
    __shared__ __bf16 hB[2][BB][LD_H];        // double-buffered hidden (bf16)
    __shared__ float  alp[BB][MAA], bet[BB][MAA], kap[BB][MAA];
    __shared__ float  phi[BB][UU];

    const int tid  = threadIdx.x;
    const int lane = tid & 31;
    const int wid  = tid >> 5;                // 0..15

    for (int i = tid; i < BB * LD_A0; i += 512) ((__bf16*)aT)[i] = (__bf16)0.0f;
    for (int i = tid; i < 2 * BB * LD_H; i += 512) ((__bf16*)hB)[i] = (__bf16)0.0f;
    for (int i = tid; i < BB * MAA; i += 512) ((float*)kap)[i] = 0.0f;

    v8f creg[8];   // persistent cell state: [js*4 + mb]
#pragma unroll
    for (int s = 0; s < 8; ++s) creg[s] = (v8f){0,0,0,0,0,0,0,0};

    __syncthreads();

    const int n0  = lane & 15;
    const int hlf = lane >> 4;

    int pr = 0;
    for (int t = 0; t < TT; ++t) {
        // ---- stage x_t into aT and X1 ----
        if (tid < BB * 3) {
            int b = tid / 3, c = tid % 3;
            float xv = strokes[((size_t)b * TT + t) * 3 + c];
            __bf16 xb = (__bf16)xv;
            aT[b][c] = xb;
            X1[((size_t)t * BB + b) * LD_X + c] = xb;
        }
        __syncthreads();

        // ---- gates: [64 x 1600] = [x|w]@Wih0^T + h@Whh0^T via WMMA ----
#pragma unroll 1
        for (int js = 0; js < 2; ++js) {
            int nu = wid + js * 16;              // hidden-unit tile 0..24
            if (nu >= 25) continue;
            v8f acc[16];                          // [mb*4 + gate]
#pragma unroll
            for (int q = 0; q < 16; ++q) acc[q] = (v8f){0,0,0,0,0,0,0,0};

            // input part: K over 288 (B-frags held across all 4 batch tiles)
            for (int ch = 0; ch < LD_A0 / 32; ++ch) {
                v16bf bf[4];
#pragma unroll
                for (int g = 0; g < 4; ++g)
                    bf[g] = load_b_frag(Wih0t, G4, ch * 32, nu * 16 + g * HH, lane);
#pragma unroll
                for (int mb = 0; mb < 4; ++mb) {
                    v16bf af = load_a_frag(&aT[mb * 16][0], LD_A0, ch * 32, lane);
#pragma unroll
                    for (int g = 0; g < 4; ++g)
                        acc[mb * 4 + g] = wmma_bf16(af, bf[g], acc[mb * 4 + g]);
                }
            }
            // recurrent part: K over 416
            for (int ch = 0; ch < LD_H / 32; ++ch) {
                v16bf bf[4];
#pragma unroll
                for (int g = 0; g < 4; ++g)
                    bf[g] = load_b_frag(Whh0t, G4, ch * 32, nu * 16 + g * HH, lane);
#pragma unroll
                for (int mb = 0; mb < 4; ++mb) {
                    v16bf af = load_a_frag(&hB[pr][mb * 16][0], LD_H, ch * 32, lane);
#pragma unroll
                    for (int g = 0; g < 4; ++g)
                        acc[mb * 4 + g] = wmma_bf16(af, bf[g], acc[mb * 4 + g]);
                }
            }

            // LSTM nonlinearity; C-layout: lane -> N, reg r -> M = r + 8*(lane>=16)
            const int j = nu * 16 + n0;
            const float bI = bsum0[j], bF = bsum0[j + HH];
            const float bG = bsum0[j + 2 * HH], bO = bsum0[j + 3 * HH];
#pragma unroll
            for (int mb = 0; mb < 4; ++mb) {
#pragma unroll
                for (int r = 0; r < 8; ++r) {
                    int b = mb * 16 + r + hlf * 8;
                    float iv = sigmf(acc[mb * 4 + 0][r] + bI);
                    float fv = sigmf(acc[mb * 4 + 1][r] + bF);
                    float gv = tanhf(acc[mb * 4 + 2][r] + bG);
                    float ov = sigmf(acc[mb * 4 + 3][r] + bO);
                    float cn = fv * creg[js * 4 + mb][r] + iv * gv;
                    creg[js * 4 + mb][r] = cn;
                    float hn = ov * tanhf(cn);
                    __bf16 hb = (__bf16)hn;
                    hB[pr ^ 1][b][j] = hb;                   // next-step A operand
                    size_t row = (size_t)t * BB + b;
                    X1[row * LD_X + 3 + j] = hb;             // layer1 input
                    F[row * LD_F + j]      = hb;             // feats[:,0:400]
                }
            }
        }
        __syncthreads();

        // ---- attention: abk = exp(h @ W_att^T + b_att) ----
        for (int job = tid; job < BB * 30; job += 512) {
            int b = job / 30, j = job % 30;
            float s = b_att[j];
            const __bf16* hp = &hB[pr ^ 1][b][0];
            const float* wr = W_att + (size_t)j * HH;
            for (int k = 0; k < HH; ++k) s += (float)hp[k] * wr[k];
            float e = __expf(s);
            if (j < 10)      alp[b][j] = e;
            else if (j < 20) bet[b][j - 10] = e;
            else             kap[b][j - 20] += e;            // monotone kappa
        }
        __syncthreads();

        // ---- phi[b][u] ----
        for (int job = tid; job < BB * UU; job += 512) {
            int b = job / UU, u = job % UU;
            float uu = (float)u, s = 0.0f;
#pragma unroll
            for (int m = 0; m < MAA; ++m) {
                float d = kap[b][m] - uu;
                s += alp[b][m] * __expf(-bet[b][m] * d * d);
            }
            phi[b][u] = s * cmask[b * UU + u];
        }
        __syncthreads();

        // ---- w[b][e] = phi @ context; feed next step + X1 w-column ----
        for (int job = tid; job < BB * EE; job += 512) {
            int b = job / EE, e = job % EE;
            float s = 0.0f;
            const float* cp = context + ((size_t)b * UU) * EE + e;
            for (int u = 0; u < UU; ++u) s += phi[b][u] * cp[(size_t)u * EE];
            __bf16 wb = (__bf16)s;
            aT[b][3 + e] = wb;
            X1[((size_t)t * BB + b) * LD_X + 403 + e] = wb;
        }
        __syncthreads();
        pr ^= 1;
    }
}

// ---------------------------------------------------------------------------
// Bulk GEMM: G[M x 1600](f32) = X[M x 672](bf16) @ Wt[672 x 1600] + bias.
// Wave job = 64 rows x 64 cols (4 row-tiles share the held B fragments ->
// 4x less A and B traffic through L2 than 16-row jobs).
// ---------------------------------------------------------------------------
__global__ void __launch_bounds__(128) gemm_gates_kernel(
    const __bf16* __restrict__ X, const __bf16* __restrict__ Wt,
    const float* __restrict__ bsum, float* __restrict__ G) {
    const int lane = threadIdx.x & 31;
    const int wid  = threadIdx.x >> 5;
    const int job  = blockIdx.x * 4 + wid;       // (M/64)*25 jobs
    const int mG   = job / 25;                   // 64-row group
    const int nQ   = job % 25;                   // 64-col group

    v8f acc[16];                                  // [mt*4 + ng]
#pragma unroll
    for (int q = 0; q < 16; ++q) acc[q] = (v8f){0,0,0,0,0,0,0,0};

    for (int ch = 0; ch < LD_X / 32; ++ch) {
        v16bf bf[4];
#pragma unroll
        for (int g = 0; g < 4; ++g)
            bf[g] = load_b_frag(Wt, G4, ch * 32, nQ * 64 + g * 16, lane);
#pragma unroll
        for (int mt = 0; mt < 4; ++mt) {
            const __bf16* Arow = X + (size_t)(mG * 4 + mt) * 16 * LD_X;
            v16bf af = load_a_frag(Arow, LD_X, ch * 32, lane);
#pragma unroll
            for (int g = 0; g < 4; ++g)
                acc[mt * 4 + g] = wmma_bf16(af, bf[g], acc[mt * 4 + g]);
        }
    }
    const int n0 = lane & 15, hlf = lane >> 4;
#pragma unroll
    for (int mt = 0; mt < 4; ++mt) {
#pragma unroll
        for (int g = 0; g < 4; ++g) {
            int col = nQ * 64 + g * 16 + n0;
            float bs = bsum[col];
#pragma unroll
            for (int r = 0; r < 8; ++r) {
                int row = (mG * 4 + mt) * 16 + r + hlf * 8;
                G[(size_t)row * G4 + col] = acc[mt * 4 + g][r] + bs;
            }
        }
    }
}

// ---------------------------------------------------------------------------
// Layers 1/2 recurrence: gates_t = G[t] (precomputed input proj + bias)
//                                + h_{t-1} @ Whh^T. nu-major persistent waves,
// Whh crosses L2 exactly once per step. Prefetches next step's G block.
// ---------------------------------------------------------------------------
__global__ void __launch_bounds__(512) recur_kernel(
    const float* __restrict__ G, const __bf16* __restrict__ Whht,
    __bf16* __restrict__ Xout, __bf16* __restrict__ F, int fcol) {

    __shared__ __bf16 hB[2][BB][LD_H];
    const int tid = threadIdx.x, lane = tid & 31, wid = tid >> 5;

    for (int i = tid; i < 2 * BB * LD_H; i += 512) ((__bf16*)hB)[i] = (__bf16)0.0f;
    v8f creg[8];
#pragma unroll
    for (int s = 0; s < 8; ++s) creg[s] = (v8f){0,0,0,0,0,0,0,0};
    __syncthreads();

    const int n0 = lane & 15, hlf = lane >> 4;

    int pr = 0;
    for (int t = 0; t < TT; ++t) {
        // prefetch next step's precomputed gates while we compute this one
        if (t + 1 < TT) {
            const char* Gn = (const char*)(G + (size_t)(t + 1) * BB * G4);
            for (size_t o = (size_t)tid * 128; o < (size_t)BB * G4 * 4; o += 512 * 128)
                __builtin_prefetch(Gn + o, 0, 1);
        }
#pragma unroll 1
        for (int js = 0; js < 2; ++js) {
            int nu = wid + js * 16;
            if (nu >= 25) continue;

            v8f acc[16];
#pragma unroll
            for (int mb = 0; mb < 4; ++mb) {      // seed with precomputed proj
#pragma unroll
                for (int g = 0; g < 4; ++g) {
#pragma unroll
                    for (int r = 0; r < 8; ++r) {
                        size_t row = (size_t)t * BB + mb * 16 + r + hlf * 8;
                        acc[mb * 4 + g][r] = G[row * G4 + g * HH + nu * 16 + n0];
                    }
                }
            }
            for (int ch = 0; ch < LD_H / 32; ++ch) {
                v16bf bf[4];
#pragma unroll
                for (int g = 0; g < 4; ++g)
                    bf[g] = load_b_frag(Whht, G4, ch * 32, nu * 16 + g * HH, lane);
#pragma unroll
                for (int mb = 0; mb < 4; ++mb) {
                    v16bf af = load_a_frag(&hB[pr][mb * 16][0], LD_H, ch * 32, lane);
#pragma unroll
                    for (int g = 0; g < 4; ++g)
                        acc[mb * 4 + g] = wmma_bf16(af, bf[g], acc[mb * 4 + g]);
                }
            }
            const int j = nu * 16 + n0;
#pragma unroll
            for (int mb = 0; mb < 4; ++mb) {
#pragma unroll
                for (int r = 0; r < 8; ++r) {
                    int b = mb * 16 + r + hlf * 8;
                    float iv = sigmf(acc[mb * 4 + 0][r]);
                    float fv = sigmf(acc[mb * 4 + 1][r]);
                    float gv = tanhf(acc[mb * 4 + 2][r]);
                    float ov = sigmf(acc[mb * 4 + 3][r]);
                    float cn = fv * creg[js * 4 + mb][r] + iv * gv;
                    creg[js * 4 + mb][r] = cn;
                    float hn = ov * tanhf(cn);
                    __bf16 hb = (__bf16)hn;
                    hB[pr ^ 1][b][j] = hb;
                    size_t row = (size_t)t * BB + b;
                    F[row * LD_F + fcol + j] = hb;
                    if (Xout) Xout[row * LD_X + 3 + j] = hb; // overwrite h column
                }
            }
        }
        __syncthreads();
        pr ^= 1;
    }
}

// ---------------------------------------------------------------------------
// Final FC: out[(b*T+t)*121 + n] = F[t*64+b, :1216] @ Wfct + bfc. Same 64x64
// wave blocking as the gates GEMM.
// ---------------------------------------------------------------------------
__global__ void __launch_bounds__(128) fc_kernel(
    const __bf16* __restrict__ F, const __bf16* __restrict__ Wt,
    const float* __restrict__ bfc, float* __restrict__ out) {
    const int lane = threadIdx.x & 31;
    const int wid  = threadIdx.x >> 5;
    const int job  = blockIdx.x * 4 + wid;       // (M/64)*2 jobs
    const int mG   = job / 2;
    const int nQ   = job % 2;

    v8f acc[16];
#pragma unroll
    for (int q = 0; q < 16; ++q) acc[q] = (v8f){0,0,0,0,0,0,0,0};

    for (int ch = 0; ch < LD_F / 32; ++ch) {
        v16bf bf[4];
#pragma unroll
        for (int g = 0; g < 4; ++g)
            bf[g] = load_b_frag(Wt, 128, ch * 32, nQ * 64 + g * 16, lane);
#pragma unroll
        for (int mt = 0; mt < 4; ++mt) {
            const __bf16* Arow = F + (size_t)(mG * 4 + mt) * 16 * LD_F;
            v16bf af = load_a_frag(Arow, LD_F, ch * 32, lane);
#pragma unroll
            for (int g = 0; g < 4; ++g)
                acc[mt * 4 + g] = wmma_bf16(af, bf[g], acc[mt * 4 + g]);
        }
    }
    const int n0 = lane & 15, hlf = lane >> 4;
#pragma unroll
    for (int mt = 0; mt < 4; ++mt) {
#pragma unroll
        for (int g = 0; g < 4; ++g) {
            int col = nQ * 64 + g * 16 + n0;
            if (col >= OUTN) continue;
            float bs = bfc[col];
#pragma unroll
            for (int r = 0; r < 8; ++r) {
                int row = (mG * 4 + mt) * 16 + r + hlf * 8;
                int t = row / BB, b = row % BB;
                out[((size_t)b * TT + t) * OUTN + col] = acc[mt * 4 + g][r] + bs;
            }
        }
    }
}

// ---------------------------------------------------------------------------
// Host-side orchestration
// ---------------------------------------------------------------------------
extern "C" void kernel_launch(void* const* d_in, const int* in_sizes, int n_in,
                              void* d_out, int out_size, void* d_ws, size_t ws_size,
                              hipStream_t stream) {
    const float* strokes = (const float*)d_in[0];
    const float* context = (const float*)d_in[1];
    const float* cmask   = (const float*)d_in[2];
    const float* W_ih0   = (const float*)d_in[3];
    const float* W_hh0   = (const float*)d_in[4];
    const float* b_ih0   = (const float*)d_in[5];
    const float* b_hh0   = (const float*)d_in[6];
    const float* W_att   = (const float*)d_in[7];
    const float* b_att   = (const float*)d_in[8];
    const float* W_ih1   = (const float*)d_in[9];
    const float* W_hh1   = (const float*)d_in[10];
    const float* b_ih1   = (const float*)d_in[11];
    const float* b_hh1   = (const float*)d_in[12];
    const float* W_ih2   = (const float*)d_in[13];
    const float* W_hh2   = (const float*)d_in[14];
    const float* b_ih2   = (const float*)d_in[15];
    const float* b_hh2   = (const float*)d_in[16];
    const float* W_fc    = (const float*)d_in[17];
    const float* b_fc    = (const float*)d_in[18];
    float* out = (float*)d_out;

    char* base = (char*)d_ws;
    size_t off = 0;
    auto take = [&](size_t bytes) -> void* {
        void* p = base + off;
        off = (off + bytes + 255) & ~(size_t)255;
        return p;
    };
    __bf16* Wih0t = (__bf16*)take((size_t)LD_A0 * G4 * 2);
    __bf16* Whh0t = (__bf16*)take((size_t)LD_H  * G4 * 2);
    __bf16* Wih1t = (__bf16*)take((size_t)LD_X  * G4 * 2);
    __bf16* Whh1t = (__bf16*)take((size_t)LD_H  * G4 * 2);
    __bf16* Wih2t = (__bf16*)take((size_t)LD_X  * G4 * 2);
    __bf16* Whh2t = (__bf16*)take((size_t)LD_H  * G4 * 2);
    __bf16* Wfct  = (__bf16*)take((size_t)LD_F  * 128 * 2);
    float*  bsum0 = (float*)take(G4 * 4);
    float*  bsum1 = (float*)take(G4 * 4);
    float*  bsum2 = (float*)take(G4 * 4);
    float*  bfc   = (float*)take(128 * 4);
    __bf16* X1    = (__bf16*)take((size_t)MROWS * LD_X * 2);
    __bf16* F     = (__bf16*)take((size_t)MROWS * LD_F * 2);
    float*  G     = (float*)take((size_t)MROWS * G4 * 4);
    if (off > ws_size) return;   // workspace too small: do nothing

    auto tgrid = [](int n) { return (n + 255) / 256; };

    // 1) weight transposition / bf16 conversion / padding
    transpose_bf16_kernel<<<tgrid(LD_A0 * G4), 256, 0, stream>>>(W_ih0, Wih0t, G4, 3 + EE, LD_A0, G4);
    transpose_bf16_kernel<<<tgrid(LD_H  * G4), 256, 0, stream>>>(W_hh0, Whh0t, G4, HH, LD_H, G4);
    transpose_bf16_kernel<<<tgrid(LD_X  * G4), 256, 0, stream>>>(W_ih1, Wih1t, G4, 3 + EE + HH, LD_X, G4);
    transpose_bf16_kernel<<<tgrid(LD_H  * G4), 256, 0, stream>>>(W_hh1, Whh1t, G4, HH, LD_H, G4);
    transpose_bf16_kernel<<<tgrid(LD_X  * G4), 256, 0, stream>>>(W_ih2, Wih2t, G4, 3 + EE + HH, LD_X, G4);
    transpose_bf16_kernel<<<tgrid(LD_H  * G4), 256, 0, stream>>>(W_hh2, Whh2t, G4, HH, LD_H, G4);
    transpose_bf16_kernel<<<tgrid(LD_F  * 128), 256, 0, stream>>>(W_fc, Wfct, OUTN, 3 * HH, LD_F, 128);

    // 2) fused biases (b_ih + b_hh)
    bias_fuse_kernel<<<tgrid(G4), 256, 0, stream>>>(b_ih0, b_hh0, bsum0, G4, G4);
    bias_fuse_kernel<<<tgrid(G4), 256, 0, stream>>>(b_ih1, b_hh1, bsum1, G4, G4);
    bias_fuse_kernel<<<tgrid(G4), 256, 0, stream>>>(b_ih2, b_hh2, bsum2, G4, G4);
    bias_fuse_kernel<<<1, 256, 0, stream>>>(b_fc, nullptr, bfc, OUTN, 128);

    // 3) zero K-padding columns of X1 and F
    zeropad_kernel<<<512, 256, 0, stream>>>(X1, LD_X, 659, LD_X, MROWS);
    zeropad_kernel<<<512, 256, 0, stream>>>(F, LD_F, 1200, LD_F, MROWS);

    // 4) layer-0 sequential scan with attention (persistent, WMMA gates)
    scan0_kernel<<<1, 512, 0, stream>>>(strokes, context, cmask,
                                        Wih0t, Whh0t, bsum0, W_att, b_att, X1, F);

    // 5) layer-1: bulk input projection (WMMA GEMM) then recurrence
    gemm_gates_kernel<<<(MROWS / 64) * 25 / 4, 128, 0, stream>>>(X1, Wih1t, bsum1, G);
    recur_kernel<<<1, 512, 0, stream>>>(G, Whh1t, X1, F, HH);

    // 6) layer-2: same, X1's h-column now holds h1 (x/w columns unchanged)
    gemm_gates_kernel<<<(MROWS / 64) * 25 / 4, 128, 0, stream>>>(X1, Wih2t, bsum2, G);
    recur_kernel<<<1, 512, 0, stream>>>(G, Whh2t, nullptr, F, 2 * HH);

    // 7) final FC straight into d_out with [B,T,OUT] mapping
    fc_kernel<<<(MROWS / 64) * 2 / 4, 128, 0, stream>>>(F, Wfct, bfc, out);
}
